// KVCacheAttention_42099269435815
// MI455X (gfx1250) — compile-verified, matrix operand order checked by reading, not run on hardware
//
#include <hip/hip_runtime.h>

// Problem constants
#define HD      128
#define NH      16
#define TQ      32
#define DMODEL  2048
#define SCACHE  4096
#define NCHUNK  129      // (SCACHE + TQ) / 32
#define NPART   4        // flash-decoding partitions over the key axis
#define MROWS   512      // B*T
#define KPITCH  132      // LDS row pitch (floats): 4-bank skew per row
#define SCALE_F 0.08838834764831845f  // 1/sqrt(128)

typedef __attribute__((ext_vector_type(16))) _Float16 v16h;
typedef __attribute__((ext_vector_type(8)))  _Float16 v8h;
typedef __attribute__((ext_vector_type(8)))  float    v8f;
typedef __attribute__((ext_vector_type(4)))  float    f32x4;

// ---- CDNA5 async copy: global -> LDS, tracked with ASYNCcnt -------------
__device__ __forceinline__ void async_ld16(unsigned lds_off, const void* g) {
  // per-lane: LDS[lds_off] = MEM[g], 16 bytes (GLOBAL_LOAD_ASYNC_TO_LDS_B128)
  asm volatile("global_load_async_to_lds_b128 %0, %1, off"
               :: "v"(lds_off), "v"(g)
               : "memory");
}
__device__ __forceinline__ void wait_async0() {
  asm volatile("s_wait_asynccnt 0" ::: "memory");
}
__device__ __forceinline__ unsigned lds_addr32(const void* p) {
  // addrspace(3) offset lives in the low 32 bits of the generic pointer
  return (unsigned)(uintptr_t)p;
}

// Build one 16-element f16 A/B fragment slice for this lane from fp32 memory.
// Caller passes p already offset by (kbase + 8*half); elements 0..7 = p[0..7],
// elements 8..15 = p[16..23]  (16-bit-matrix K-run layout, ISA 7.12.2).
__device__ __forceinline__ v16h cvt_frag_f32(const float* p) {
  f32x4 a0 = *(const f32x4*)(p + 0);
  f32x4 a1 = *(const f32x4*)(p + 4);
  f32x4 b0 = *(const f32x4*)(p + 16);
  f32x4 b1 = *(const f32x4*)(p + 20);
  v16h r;
#pragma unroll
  for (int i = 0; i < 4; ++i) {
    r[i]      = (_Float16)a0[i];
    r[i + 4]  = (_Float16)a1[i];
    r[i + 8]  = (_Float16)b0[i];
    r[i + 12] = (_Float16)b1[i];
  }
  return r;
}

// C = A(MxK) * W(NxK)^T + bias. One wave owns a 16x64 output tile (4 WMMA
// accumulators), K-loop step 32. fp32 in, f16 WMMA, f32 accumulate.
// Weights are L2-resident (192MB L2), so no LDS tiling needed.
// mode 0: scatter QKV into per-head (B,H,T,DK) q/k/v buffers.
// mode 1: plain row-major MxN store to o0.
__global__ __launch_bounds__(256)
void gemm_xwT_f16(const float* __restrict__ A, const float* __restrict__ W,
                  const float* __restrict__ bias, int K, int N, int mode,
                  float* __restrict__ o0, float* __restrict__ o1,
                  float* __restrict__ o2) {
  const int wid  = blockIdx.x * (blockDim.x >> 5) + (threadIdx.x >> 5);
  const int lane = threadIdx.x & 31;
  const int half = lane >> 4;
  const int lm   = lane & 15;
  const int ngroups = N >> 6;               // 64-wide column groups
  const int m0 = (wid / ngroups) << 4;
  const int n0 = (wid % ngroups) << 6;

  const float* arow = A + (size_t)(m0 + lm) * K;
  const float* w0   = W + (size_t)(n0 + lm) * K;

  v8f acc[4] = {};
  for (int kb = 0; kb < K; kb += 32) {
    const int ka = kb + 8 * half;
    v16h af = cvt_frag_f32(arow + ka);
#pragma unroll
    for (int j = 0; j < 4; ++j) {
      v16h bf = cvt_frag_f32(w0 + (size_t)(j * 16) * K + ka);
      acc[j] = __builtin_amdgcn_wmma_f32_16x16x32_f16(
          false, af, false, bf, (short)0, acc[j], false, false);
    }
  }

#pragma unroll
  for (int j = 0; j < 4; ++j) {
    const int n = n0 + j * 16 + lm;
    const float bv = bias[n];
#pragma unroll
    for (int r = 0; r < 8; ++r) {
      const int m = m0 + r + 8 * half;      // C layout: row = r + 8*(lane/16)
      const float v = acc[j][r] + bv;
      if (mode == 0) {
        const int sel = n >> 11;            // 0:q 1:k 2:v
        const int nn  = n & 2047;
        const int h   = nn >> 7;
        const int dk  = nn & 127;
        const int b   = m >> 5;
        const int t   = m & 31;
        float* o = (sel == 0) ? o0 : ((sel == 1) ? o1 : o2);
        o[(((size_t)(b * NH + h)) * TQ + t) * HD + dk] = v;
      } else {
        o0[(size_t)m * N + n] = v;
      }
    }
  }
}

// Stage one 32x128 fp32 K/V chunk (keys [s0,s0+32)) into LDS with async
// copies; rows are padded to KPITCH floats for bank skew. 16 b128 ops/lane.
__device__ __forceinline__ void stage_chunk(const float* __restrict__ cache,
                                            const float* __restrict__ newbuf,
                                            int bh, int s0, unsigned lds_base,
                                            int tid) {
#pragma unroll 4
  for (int i = tid; i < 1024; i += 64) {
    const int row = i >> 5, c4 = i & 31;
    const int s = s0 + row;
    const float* g = (s < SCACHE)
        ? (cache + ((size_t)bh * SCACHE + s) * HD + c4 * 4)
        : (newbuf + ((size_t)bh * TQ + (s - SCACHE)) * HD + c4 * 4);
    async_ld16(lds_base + (unsigned)(row * KPITCH + c4 * 4) * 4u, g);
  }
}

// Flash-decoding partial attention. Workgroup = 2 waves = one (b,h) x one key
// partition; wave w owns Q rows [16w,16w+16). Software pipeline: V chunks are
// double-buffered, K restaged right after its last reader (scores), so the
// next chunk's async copies overlap scores/softmax/PV of the current chunk.
__global__ __launch_bounds__(64)
void attn_partial(const float* __restrict__ qbuf, const float* __restrict__ kbuf,
                  const float* __restrict__ vbuf,
                  const float* __restrict__ kcache,
                  const float* __restrict__ vcache,
                  float* __restrict__ opart, float* __restrict__ mpart,
                  float* __restrict__ lpart) {
  __shared__ float sK[32 * KPITCH];         // 16.5 KB
  __shared__ float sV[2][32 * KPITCH];      // 33   KB (double buffered)
  __shared__ _Float16 sP[2][16 * 32];       //  2   KB (per-wave P bounce)

  const int bh   = blockIdx.x & 255;        // b*16 + h
  const int part = blockIdx.x >> 8;
  const int tid  = threadIdx.x;
  const int wave = tid >> 5;                // q-row block
  const int lane = tid & 31;
  const int half = lane >> 4;
  const int lm   = lane & 15;

  const int c0 = (NCHUNK * part) / NPART;
  const int c1 = (NCHUNK * (part + 1)) / NPART;

  const unsigned sK_base  = lds_addr32((const void*)sK);
  const unsigned sV_base0 = lds_addr32((const void*)sV[0]);
  const unsigned sV_base1 = lds_addr32((const void*)sV[1]);

  // prologue: stage first chunk
  stage_chunk(kcache, kbuf, bh, c0 * 32, sK_base, tid);
  stage_chunk(vcache, vbuf, bh, c0 * 32, sV_base0, tid);

  // Q fragments for this wave's 16 rows, all 128 dk (4 K-slices of 32)
  const float* qrow = qbuf + ((size_t)bh * TQ + wave * 16 + lm) * HD;
  v16h aq[4];
#pragma unroll
  for (int ks = 0; ks < 4; ++ks) aq[ks] = cvt_frag_f32(qrow + ks * 32 + 8 * half);

  v8f accO[8] = {};
  float mI[8], lI[8];
#pragma unroll
  for (int r = 0; r < 8; ++r) { mI[r] = -1e30f; lI[r] = 0.f; }

  for (int c = c0; c < c1; ++c) {
    const int vb = (c - c0) & 1;
    wait_async0();                          // this wave's staged data arrived
    __syncthreads();                        // ... and the other wave's too

    // prefetch next V chunk into the other buffer (last read in the previous iteration,
    // protected by the barrier above); overlaps all compute below
    if (c + 1 < c1)
      stage_chunk(vcache, vbuf, bh, (c + 1) * 32,
                  vb ? sV_base0 : sV_base1, tid);

    // scores tile 16(q) x 32(keys) = two 16x16 WMMA accumulators
    v8f sc[2] = {};
#pragma unroll
    for (int nt = 0; nt < 2; ++nt) {
#pragma unroll
      for (int ks = 0; ks < 4; ++ks) {
        v16h bk =
            cvt_frag_f32(sK + (nt * 16 + lm) * KPITCH + ks * 32 + 8 * half);
        sc[nt] = __builtin_amdgcn_wmma_f32_16x16x32_f16(
            false, aq[ks], false, bk, (short)0, sc[nt], false, false);
      }
    }

    // sK's last reader was the loop above: restage it for the next chunk now,
    // overlapping softmax + PV. Barrier ensures both waves are done with sK.
    __syncthreads();
    if (c + 1 < c1)
      stage_chunk(kcache, kbuf, bh, (c + 1) * 32, sK_base, tid);

    // online softmax: row m = r + 8*half lives across 16 lanes of this half
    float alpha[8];
#pragma unroll
    for (int r = 0; r < 8; ++r) {
      const float s0v = sc[0][r] * SCALE_F;
      const float s1v = sc[1][r] * SCALE_F;
      float mv = fmaxf(s0v, s1v);
#pragma unroll
      for (int msk = 1; msk < 16; msk <<= 1)
        mv = fmaxf(mv, __shfl_xor(mv, msk, 32));
      const float mnew = fmaxf(mI[r], mv);
      alpha[r] = __expf(mI[r] - mnew);
      const float p0 = __expf(s0v - mnew);
      const float p1 = __expf(s1v - mnew);
      float rs = p0 + p1;
#pragma unroll
      for (int msk = 1; msk < 16; msk <<= 1) rs += __shfl_xor(rs, msk, 32);
      lI[r] = lI[r] * alpha[r] + rs;
      mI[r] = mnew;
      sc[0][r] = p0;
      sc[1][r] = p1;
    }
#pragma unroll
    for (int t = 0; t < 8; ++t)
#pragma unroll
      for (int r = 0; r < 8; ++r) accO[t][r] *= alpha[r];

    // bounce P (16x32, C-layout) through LDS into A-fragment layout (f16)
    _Float16* pbuf = sP[wave];
#pragma unroll
    for (int nt = 0; nt < 2; ++nt)
#pragma unroll
      for (int r = 0; r < 8; ++r)
        pbuf[(r + 8 * half) * 32 + nt * 16 + lm] = (_Float16)sc[nt][r];
    // same-wave LDS RAW: DS ops are in order; compiler inserts s_wait_dscnt
    v16h ap;
    {
      v8h lo = *(const v8h*)(pbuf + lm * 32 + 8 * half);
      v8h hi = *(const v8h*)(pbuf + lm * 32 + 16 + 8 * half);
      ap = __builtin_shufflevector(lo, hi, 0, 1, 2, 3, 4, 5, 6, 7, 8, 9, 10, 11,
                                   12, 13, 14, 15);
    }

    // O(16x128) += P(16x32) * V(32x128): 8 dk-tiles of WMMA
    const float* vcur = sV[vb];
#pragma unroll
    for (int dt = 0; dt < 8; ++dt) {
      const float* vcol = vcur + dt * 16 + lm;
      v16h bv;
#pragma unroll
      for (int e = 0; e < 8; ++e) {
        bv[e]     = (_Float16)vcol[(8 * half + e) * KPITCH];
        bv[e + 8] = (_Float16)vcol[(16 + 8 * half + e) * KPITCH];
      }
      accO[dt] = __builtin_amdgcn_wmma_f32_16x16x32_f16(
          false, ap, false, bv, (short)0, accO[dt], false, false);
    }
  }

  // store partials: O (f32), per-row m and l
  const size_t obase = ((size_t)part * 256 + bh) * TQ + wave * 16;
#pragma unroll
  for (int dt = 0; dt < 8; ++dt)
#pragma unroll
    for (int r = 0; r < 8; ++r)
      opart[(obase + r + 8 * half) * HD + dt * 16 + lm] = accO[dt][r];
  if (lm == 0) {
#pragma unroll
    for (int r = 0; r < 8; ++r) {
      const size_t rowi = obase + r + 8 * half;
      mpart[rowi] = mI[r];
      lpart[rowi] = lI[r];
    }
  }
}

// Merge NPART partial (O, m, l) triples; emit attn output as (B,T, H*DK)
// row-major so the final projection GEMM reads it directly.
__global__ __launch_bounds__(128)
void attn_combine(const float* __restrict__ opart, const float* __restrict__ mpart,
                  const float* __restrict__ lpart, float* __restrict__ attn) {
  const int rowg = blockIdx.x;              // bh*32 + t
  const int bh = rowg >> 5, t = rowg & 31;
  const int dk = threadIdx.x;
  float M = -1e30f;
#pragma unroll
  for (int p = 0; p < NPART; ++p)
    M = fmaxf(M, mpart[((size_t)p * 256 + bh) * TQ + t]);
  float L = 0.f, acc = 0.f;
#pragma unroll
  for (int p = 0; p < NPART; ++p) {
    const size_t ri = ((size_t)p * 256 + bh) * TQ + t;
    const float w = __expf(mpart[ri] - M);
    L += lpart[ri] * w;
    acc += opart[ri * HD + dk] * w;
  }
  const int b = bh >> 4, h = bh & 15;
  attn[((size_t)(b * TQ + t)) * DMODEL + h * HD + dk] = acc / L;
}

extern "C" void kernel_launch(void* const* d_in, const int* in_sizes, int n_in,
                              void* d_out, int out_size, void* d_ws,
                              size_t ws_size, hipStream_t stream) {
  const float* x      = (const float*)d_in[0];
  const float* kcache = (const float*)d_in[1];
  const float* vcache = (const float*)d_in[2];
  const float* qkv_w  = (const float*)d_in[3];
  const float* qkv_b  = (const float*)d_in[4];
  const float* out_w  = (const float*)d_in[5];
  const float* out_b  = (const float*)d_in[6];
  float* out = (float*)d_out;

  float* qbuf  = (float*)d_ws;                                   // 4 MB
  float* kbuf  = qbuf  + (size_t)MROWS * DMODEL;                 // 4 MB
  float* vbuf  = kbuf  + (size_t)MROWS * DMODEL;                 // 4 MB
  float* opart = vbuf  + (size_t)MROWS * DMODEL;                 // 16 MB
  float* mpart = opart + (size_t)NPART * 256 * TQ * HD;
  float* lpart = mpart + (size_t)NPART * 256 * TQ;
  float* attn  = lpart + (size_t)NPART * 256 * TQ;               // 4 MB

  // 1) fused QKV projection: (512x2048) @ (6144x2048)^T, scatter per head
  {
    const int waves = (MROWS / 16) * ((3 * DMODEL) / 64);        // 3072
    gemm_xwT_f16<<<waves / 8, 256, 0, stream>>>(x, qkv_w, qkv_b, DMODEL,
                                                3 * DMODEL, 0, qbuf, kbuf, vbuf);
  }
  // 2) flash-decoding attention partials: 256 (b,h) x 4 key partitions
  attn_partial<<<256 * NPART, 64, 0, stream>>>(qbuf, kbuf, vbuf, kcache, vcache,
                                               opart, mpart, lpart);
  // 3) merge partitions
  attn_combine<<<256 * TQ, 128, 0, stream>>>(opart, mpart, lpart, attn);
  // 4) output projection: (512x2048) @ (2048x2048)^T + bias
  {
    const int waves = (MROWS / 16) * (DMODEL / 64);              // 1024
    gemm_xwT_f16<<<waves / 8, 256, 0, stream>>>(attn, out_w, out_b, DMODEL,
                                                DMODEL, 1, out, nullptr,
                                                nullptr);
  }
}
